// ChannelAttentionModule_32444182954275
// MI455X (gfx1250) — matrix operand order repaired and checked by reading, not compiled, not run on hardware
//
#include <hip/hip_runtime.h>
#include <hip/hip_bf16.h>

// ---------------------------------------------------------------------------
// Channel attention:  S = Q Q^T (per batch), A = softmax(S), out = g*(A Q) + x
//   B=16, C=512, N=H*W=4096
// bf16 WMMA (v_wmma_f32_16x16x32_bf16) for both GEMMs, f32 accumulate.
// LDS-staged mainloop with CDNA5 async global->LDS copies, double-buffered.
// ---------------------------------------------------------------------------

#define BATCH 16
#define CCH   512
#define NHW   4096

#define ROWE   72              // padded LDS row length in bf16 elems (144 B, 16B-aligned)
#define TILE_E (128 * ROWE)    // one LDS tile (128 rows)

typedef __attribute__((ext_vector_type(16))) __bf16 v16bf;
typedef __attribute__((ext_vector_type(8)))  float  v8f;

union FragU {
    struct { uint4 lo, hi; } u;
    v16bf v;
};

// ---------------- CDNA5 async global->LDS copy (16 B per lane) -------------
typedef int v4i __attribute__((__vector_size__(16)));
typedef __attribute__((address_space(1))) v4i* g1v4_t;   // global (prints __device__)
typedef __attribute__((address_space(3))) v4i* l3v4_t;   // LDS    (prints __shared__)

__device__ __forceinline__ void async_copy16(const __hip_bfloat16* gsrc,
                                             __hip_bfloat16* lds_dst) {
#if __has_builtin(__builtin_amdgcn_global_load_async_to_lds_b128)
    __builtin_amdgcn_global_load_async_to_lds_b128(
        (g1v4_t)(unsigned long long)(size_t)gsrc,
        (l3v4_t)(unsigned int)(size_t)lds_dst, 0, 0);
#else
    unsigned int lo = (unsigned int)(size_t)lds_dst;
    asm volatile("global_load_async_to_lds_b128 %0, %1, off"
                 :: "v"(lo), "v"(gsrc) : "memory");
#endif
}

#if __has_builtin(__builtin_amdgcn_s_wait_asynccnt)
#define WAIT_ASYNC(n) __builtin_amdgcn_s_wait_asynccnt(n)
#else
#define WAIT_ASYNC(n) asm volatile("s_wait_asynccnt " #n ::: "memory")
#endif

// ---------------- LDS fragment loads (ds_load_b128 x2) ---------------------
// A fragment: 16x32 bf16.
// lanes 0-15 : M=lane,    VGPR0-3 = K[k..k+7],   VGPR4-7 = K[k+16..k+23]
// lanes16-31 : M=lane-16, VGPR0-3 = K[k+8..k+15],VGPR4-7 = K[k+24..k+31]
__device__ __forceinline__ v16bf lds_frag_a(const __hip_bfloat16* tile,
                                            int row0, int ko, int lane) {
    int r  = row0 + (lane & 15);
    int kk = ko + ((lane >> 4) << 3);
    const uint4* p = reinterpret_cast<const uint4*>(tile + r * ROWE + kk);
    FragU f;
    f.u.lo = p[0];
    f.u.hi = p[2];          // +16 elems = +32 B
    return f.v;
}

// B fragment: 32x16 bf16 (tile rows = output columns, "NT" storage).
// lanes 0-15 : N=lane,    K[k..k+7],K[k+16..k+23] ; lanes16-31 : N=lane-16, K[k+8..k+15],K[k+24..k+31]
__device__ __forceinline__ v16bf lds_frag_b(const __hip_bfloat16* tile,
                                            int col0, int ko, int lane) {
    int c  = col0 + (lane & 15);
    int kk = ko + ((lane >> 4) << 3);
    const uint4* p = reinterpret_cast<const uint4*>(tile + c * ROWE + kk);
    FragU f;
    f.u.lo = p[0];
    f.u.hi = p[2];
    return f.v;
}

// ---------------- async stage: one 128x64 bf16 tile into LDS ---------------
// 256 threads; thread t copies 64 B (half a 128 B tile row) as 4 async b128.
__device__ __forceinline__ void stage_tile(const __hip_bfloat16* g, int ld,
                                           int base_rc, int k0,
                                           __hip_bfloat16* s, int t) {
    int row = t >> 1;
    int seg = (t & 1) << 5;                       // 0 or 32 elems
    const __hip_bfloat16* gr = g + (size_t)(base_rc + row) * ld + k0 + seg;
    __hip_bfloat16* lr = s + row * ROWE + seg;
#pragma unroll
    for (int j = 0; j < 4; ++j)
        async_copy16(gr + j * 8, lr + j * 8);
}

// ---------------- shared GEMM mainloop (NT, 128x128 block tile) ------------
template <int KTOT>
__device__ __forceinline__ void gemm_mainloop(const __hip_bfloat16* __restrict__ A,
                                              int lda,
                                              const __hip_bfloat16* __restrict__ B,
                                              int ldb,
                                              int row0, int col0,
                                              __hip_bfloat16* sA,
                                              __hip_bfloat16* sB,
                                              v8f (&acc)[4][2]) {
    const int t     = threadIdx.x;
    const int lane  = t & 31;
    const int w     = t >> 5;
    const int wrowL = (w >> 2) * 64;              // local wave tile: 64 x 32
    const int wcolL = (w & 3) * 32;

#pragma unroll
    for (int mt = 0; mt < 4; ++mt)
#pragma unroll
        for (int nt = 0; nt < 2; ++nt)
            acc[mt][nt] = (v8f){0.f, 0.f, 0.f, 0.f, 0.f, 0.f, 0.f, 0.f};

    stage_tile(A, lda, row0, 0, sA, t);
    stage_tile(B, ldb, col0, 0, sB, t);

    for (int k0 = 0; k0 < KTOT; k0 += 64) {
        const int buf = (k0 >> 6) & 1;
        if (k0 + 64 < KTOT) {
            stage_tile(A, lda, row0, k0 + 64, sA + (buf ^ 1) * TILE_E, t);
            stage_tile(B, ldb, col0, k0 + 64, sB + (buf ^ 1) * TILE_E, t);
            WAIT_ASYNC(8);                        // current buffer landed (in-order)
        } else {
            WAIT_ASYNC(0);
        }
        __syncthreads();

        const __hip_bfloat16* tA = sA + buf * TILE_E;
        const __hip_bfloat16* tB = sB + buf * TILE_E;
#pragma unroll
        for (int s = 0; s < 2; ++s) {             // two 32-wide K slices
            const int ko = s * 32;
            v16bf af[4], bfr[2];
#pragma unroll
            for (int mt = 0; mt < 4; ++mt)
                af[mt] = lds_frag_a(tA, wrowL + mt * 16, ko, lane);
#pragma unroll
            for (int nt = 0; nt < 2; ++nt)
                bfr[nt] = lds_frag_b(tB, wcolL + nt * 16, ko, lane);
#pragma unroll
            for (int mt = 0; mt < 4; ++mt)
#pragma unroll
                for (int nt = 0; nt < 2; ++nt)
                    acc[mt][nt] = __builtin_amdgcn_wmma_f32_16x16x32_bf16(
                        false, af[mt], false, bfr[nt], (short)0, acc[mt][nt],
                        false, false);
        }
        __syncthreads();                          // all reads done before re-stage
    }
}

// ---------------------------------------------------------------------------
// Kernel 1: f32 -> bf16 copy of Q, plus bf16 Q^T (LDS-tiled transpose).
// grid (C/32, N/32, B), block 256 (32x8)
// ---------------------------------------------------------------------------
__global__ void __launch_bounds__(256)
prep_kernel(const float* __restrict__ x,
            __hip_bfloat16* __restrict__ qb,
            __hip_bfloat16* __restrict__ qt) {
    __shared__ __hip_bfloat16 tile[32][33];
    const int tx = threadIdx.x & 31;
    const int ty = threadIdx.x >> 5;
    const int c0 = blockIdx.x * 32;
    const int n0 = blockIdx.y * 32;
    const int b  = blockIdx.z;
    const size_t base = (size_t)b * CCH * NHW;

#pragma unroll
    for (int i = 0; i < 4; ++i) {
        int r = ty + i * 8;
        float v = x[base + (size_t)(c0 + r) * NHW + (n0 + tx)];
        __hip_bfloat16 h = __float2bfloat16(v);
        qb[base + (size_t)(c0 + r) * NHW + (n0 + tx)] = h;
        tile[r][tx] = h;
    }
    __syncthreads();
#pragma unroll
    for (int i = 0; i < 4; ++i) {
        int r = ty + i * 8;
        qt[(size_t)b * NHW * CCH + (size_t)(n0 + r) * CCH + (c0 + tx)] = tile[tx][r];
    }
}

// ---------------------------------------------------------------------------
// Kernel 2: scores[b] = Qb Qb^T   (M=N=512, K=4096)
// ---------------------------------------------------------------------------
__global__ void __launch_bounds__(256)
gemm1_kernel(const __hip_bfloat16* __restrict__ qb,
             float* __restrict__ scores) {
    __shared__ __hip_bfloat16 sA[2 * TILE_E];
    __shared__ __hip_bfloat16 sB[2 * TILE_E];

    const int lane = threadIdx.x & 31;
    const int w    = threadIdx.x >> 5;
    const int b    = blockIdx.z;
    const int row0 = blockIdx.x * 128;
    const int col0 = blockIdx.y * 128;

    const __hip_bfloat16* Q = qb + (size_t)b * CCH * NHW;

    v8f acc[4][2];
    gemm_mainloop<NHW>(Q, NHW, Q, NHW, row0, col0, sA, sB, acc);

    float* S = scores + (size_t)b * CCH * CCH;
    const int wrow  = row0 + (w >> 2) * 64;
    const int wcol  = col0 + (w & 3) * 32;
    const int rbase = (lane >> 4) << 3;
    const int cidx  = lane & 15;
#pragma unroll
    for (int mt = 0; mt < 4; ++mt)
#pragma unroll
        for (int nt = 0; nt < 2; ++nt)
#pragma unroll
            for (int i = 0; i < 8; ++i) {
                int r = wrow + mt * 16 + rbase + i;
                int c = wcol + nt * 16 + cidx;
                S[(size_t)r * CCH + c] = acc[mt][nt][i];
            }
}

// ---------------------------------------------------------------------------
// Kernel 3: row softmax over 512, write bf16 attn. One block per row.
// ---------------------------------------------------------------------------
__global__ void __launch_bounds__(256)
softmax_kernel(const float* __restrict__ scores,
               __hip_bfloat16* __restrict__ attn) {
    const int row = blockIdx.x;
    const int t   = threadIdx.x;
    const float* s = scores + (size_t)row * CCH;
    __shared__ float red[256];

    float a0 = s[t];
    float a1 = s[t + 256];

    red[t] = fmaxf(a0, a1);
    __syncthreads();
    for (int off = 128; off > 0; off >>= 1) {
        if (t < off) red[t] = fmaxf(red[t], red[t + off]);
        __syncthreads();
    }
    float m = red[0];
    __syncthreads();

    float e0 = __expf(a0 - m);
    float e1 = __expf(a1 - m);
    red[t] = e0 + e1;
    __syncthreads();
    for (int off = 128; off > 0; off >>= 1) {
        if (t < off) red[t] += red[t + off];
        __syncthreads();
    }
    float inv = 1.0f / red[0];

    attn[(size_t)row * CCH + t]       = __float2bfloat16(e0 * inv);
    attn[(size_t)row * CCH + t + 256] = __float2bfloat16(e1 * inv);
}

// ---------------------------------------------------------------------------
// Kernel 4: out = gamma * (A Q) + x   (M=512, N=4096, K=512)
//   A = attn (ld=512), B = Q^T rows (qt, ld=512)
// ---------------------------------------------------------------------------
__global__ void __launch_bounds__(256)
gemm2_kernel(const __hip_bfloat16* __restrict__ attn,
             const __hip_bfloat16* __restrict__ qt,
             const float* __restrict__ x,
             const float* __restrict__ gamma,
             float* __restrict__ out) {
    __shared__ __hip_bfloat16 sA[2 * TILE_E];
    __shared__ __hip_bfloat16 sB[2 * TILE_E];

    const int lane = threadIdx.x & 31;
    const int w    = threadIdx.x >> 5;
    const int b    = blockIdx.z;
    const int row0 = blockIdx.x * 128;
    const int col0 = blockIdx.y * 128;

    const __hip_bfloat16* A  = attn + (size_t)b * CCH * CCH;
    const __hip_bfloat16* Bt = qt   + (size_t)b * NHW * CCH;

    v8f acc[4][2];
    gemm_mainloop<CCH>(A, CCH, Bt, CCH, row0, col0, sA, sB, acc);

    const float g = gamma[0];
    const float* xb = x   + (size_t)b * CCH * NHW;
    float*       ob = out + (size_t)b * CCH * NHW;
    const int wrow  = row0 + (w >> 2) * 64;
    const int wcol  = col0 + (w & 3) * 32;
    const int rbase = (lane >> 4) << 3;
    const int cidx  = lane & 15;
#pragma unroll
    for (int mt = 0; mt < 4; ++mt)
#pragma unroll
        for (int nt = 0; nt < 2; ++nt)
#pragma unroll
            for (int i = 0; i < 8; ++i) {
                int r = wrow + mt * 16 + rbase + i;
                int c = wcol + nt * 16 + cidx;
                size_t idx = (size_t)r * NHW + c;
                ob[idx] = g * acc[mt][nt][i] + xb[idx];
            }
}

// ---------------------------------------------------------------------------
extern "C" void kernel_launch(void* const* d_in, const int* in_sizes, int n_in,
                              void* d_out, int out_size, void* d_ws, size_t ws_size,
                              hipStream_t stream) {
    const float* x     = (const float*)d_in[0];
    const float* gamma = (const float*)d_in[1];
    float*       out   = (float*)d_out;

    char* ws = (char*)d_ws;
    const size_t qb_bytes = (size_t)BATCH * CCH * NHW * sizeof(__hip_bfloat16); // 64 MiB
    const size_t qt_bytes = qb_bytes;                                            // 64 MiB
    const size_t sc_bytes = (size_t)BATCH * CCH * CCH * sizeof(float);           // 16 MiB
    __hip_bfloat16* qb = (__hip_bfloat16*)(ws);
    __hip_bfloat16* qt = (__hip_bfloat16*)(ws + qb_bytes);
    float*          sc = (float*)(ws + qb_bytes + qt_bytes);
    __hip_bfloat16* at = (__hip_bfloat16*)(ws + qb_bytes + qt_bytes + sc_bytes);

    prep_kernel<<<dim3(CCH / 32, NHW / 32, BATCH), 256, 0, stream>>>(x, qb, qt);
    gemm1_kernel<<<dim3(CCH / 128, CCH / 128, BATCH), 256, 0, stream>>>(qb, sc);
    softmax_kernel<<<dim3(BATCH * CCH), 256, 0, stream>>>(sc, at);
    gemm2_kernel<<<dim3(CCH / 128, NHW / 128, BATCH), 256, 0, stream>>>(at, qt, x, gamma, out);
}
